// MultiModalMambaKANEncoderMaskOnly_19164144075114
// MI455X (gfx1250) — matrix operand order repaired and checked by reading, not compiled, not run on hardware
//
#include <hip/hip_runtime.h>
#include <hip/hip_bf16.h>
#include <stdint.h>
#include <stddef.h>

// ---------------------------------------------------------------------------
// Model constants (from reference)
// ---------------------------------------------------------------------------
#define B_SZ     4
#define T_SEQ    8192
#define C_IN     8
#define C_OUT    8
#define AUX      16
#define D_MODEL  512
#define D_STATE  128
#define HEADDIM  64
#define D_INNER  1024
#define NH       16
#define N_LAYERS 4
#define CHUNK    64
#define NCHUNK   16          // N_TOK / CHUNK
#define PATCH    8
#define PRE_K    5
#define CONV_K   4
#define CONV_CH  1280        // D_INNER + 2*D_STATE
#define PROJ     2320        // 2*D_INNER + 2*D_STATE + NH
#define PROJ_PAD 2368        // padded to multiple of 64
#define N_TOK    1024
#define M_ROWS   4096        // B_SZ * N_TOK

typedef __bf16 bf16;
typedef __attribute__((ext_vector_type(16))) __bf16 v16bf;
typedef __attribute__((ext_vector_type(8)))  float  v8f;

#define DEV __device__ __forceinline__

DEV float silu_f(float x)      { return x / (1.0f + __expf(-x)); }
DEV float softplus_f(float x)  { return x > 20.0f ? x : log1pf(__expf(x)); }

// ---------------------------------------------------------------------------
// WMMA wrapper: D = A(16x32 bf16) * B(32x16 bf16) + C(f32)
// ---------------------------------------------------------------------------
DEV v8f wmma32(v16bf a, v16bf b, v8f c) {
#if defined(__gfx1250__)
  return __builtin_amdgcn_wmma_f32_16x16x32_bf16(
      /*neg_a=*/false, a, /*neg_b=*/false, b,
      /*c_mod=*/(short)0, c, /*reuse_a=*/false, /*reuse_b=*/false);
#else
  (void)a; (void)b;
  return c;
#endif
}

// A-fragment from LDS tile (row-major, leading dim LD), 16 rows x 32 K.
// Contiguous k order (same as frag_b): lanes 0-15 -> M=lane, K 0..15; lanes 16-31 -> K 16..31.
template <int LD>
DEV v16bf frag_a(const bf16* base, int row0, int kbase, int lane) {
  int m = lane & 15, hi = lane >> 4;
  const bf16* r = base + (row0 + m) * LD + kbase;
  v16bf a;
#pragma unroll
  for (int v = 0; v < 8; ++v) {
    int k0 = hi * 16 + v * 2;
    a[2 * v]     = r[k0];
    a[2 * v + 1] = r[k0 + 1];
  }
  return a;
}

// B-fragment from LDS tile (K-major rows, leading dim LD), 32 K x 16 N.
// Lanes 0-15 -> N=lane holding K 0..15; lanes 16-31 -> K 16..31.
template <int LD>
DEV v16bf frag_b(const bf16* base, int kbase, int col0, int lane) {
  int n = lane & 15, ks = lane >> 4;
  v16bf b;
#pragma unroll
  for (int v = 0; v < 8; ++v) {
    b[2 * v]     = base[(kbase + ks * 16 + 2 * v)     * LD + col0 + n];
    b[2 * v + 1] = base[(kbase + ks * 16 + 2 * v + 1) * LD + col0 + n];
  }
  return b;
}

// ---------------------------------------------------------------------------
// Generic tiled bf16 GEMM: C[M,N](f32) = A[M,K](bf16) * B[K,N](bf16) (+bias)(+resid)
// Block 256 thr = 8 waves; block tile 128x64; wave tile 32x32; K step 32.
// K must be a multiple of 32; ldb must cover the 64-wide N tile (pre-padded).
// ---------------------------------------------------------------------------
#define GEMM_FLAG_BIAS  1
#define GEMM_FLAG_RESID 2

__global__ __launch_bounds__(256) void gemm_bf16_wmma(
    const bf16* __restrict__ A, const bf16* __restrict__ B, float* __restrict__ C,
    const float* __restrict__ bias, const float* __restrict__ resid,
    int M, int N, int K, int lda, int ldb, int ldc, int flags) {
  __shared__ bf16 As[128][40];   // 32 K + pad (row = 80B, 16B aligned)
  __shared__ bf16 Bs[32][72];    // 64 N + pad (row = 144B, 16B aligned)

  const int tid  = threadIdx.x;
  const int lane = tid & 31;
  const int wave = tid >> 5;
  const int wm = wave >> 1, wn = wave & 1;
  const int bm = blockIdx.y * 128, bn = blockIdx.x * 64;

  v8f acc[2][2];
#pragma unroll
  for (int i = 0; i < 2; ++i)
#pragma unroll
    for (int j = 0; j < 2; ++j)
#pragma unroll
      for (int r = 0; r < 8; ++r) acc[i][j][r] = 0.0f;

  const int ar   = tid >> 1;        // A row within tile (0..127)
  const int ahalf = tid & 1;        // which 16-wide K half
  const int br   = tid >> 3;        // B row within tile (0..31)
  const int bseg = tid & 7;         // 8-wide N segment

  for (int k0 = 0; k0 < K; k0 += 32) {
    // cooperative A load: 128x32 bf16, 16 elems (2x uint4) per thread
    {
      const uint4* src = reinterpret_cast<const uint4*>(
          A + (size_t)(bm + ar) * lda + k0 + ahalf * 16);
      uint4* dst = reinterpret_cast<uint4*>(&As[ar][ahalf * 16]);
      dst[0] = src[0];
      dst[1] = src[1];
    }
    // cooperative B load: 32x64 bf16, 8 elems (1x uint4) per thread
    {
      const uint4* src = reinterpret_cast<const uint4*>(
          B + (size_t)(k0 + br) * ldb + bn + bseg * 8);
      *reinterpret_cast<uint4*>(&Bs[br][bseg * 8]) = src[0];
    }
    if (k0 + 32 < K) {  // CDNA5 global_prefetch_b8 hint for next K tile
      __builtin_prefetch(A + (size_t)(bm + ar) * lda + k0 + 32, 0, 1);
      __builtin_prefetch(B + (size_t)(k0 + 32 + br) * ldb + bn, 0, 1);
    }
    __syncthreads();

    v16bf a0 = frag_a<40>(&As[0][0], wm * 32, 0, lane);
    v16bf a1 = frag_a<40>(&As[0][0], wm * 32 + 16, 0, lane);
    v16bf b0 = frag_b<72>(&Bs[0][0], 0, wn * 32, lane);
    v16bf b1 = frag_b<72>(&Bs[0][0], 0, wn * 32 + 16, lane);
    acc[0][0] = wmma32(a0, b0, acc[0][0]);
    acc[0][1] = wmma32(a0, b1, acc[0][1]);
    acc[1][0] = wmma32(a1, b0, acc[1][0]);
    acc[1][1] = wmma32(a1, b1, acc[1][1]);
    __syncthreads();
  }

  const int hi = lane >> 4, lo = lane & 15;
#pragma unroll
  for (int mi = 0; mi < 2; ++mi)
#pragma unroll
    for (int ni = 0; ni < 2; ++ni)
#pragma unroll
      for (int r = 0; r < 8; ++r) {
        int row = bm + wm * 32 + mi * 16 + hi * 8 + r;
        int col = bn + wn * 32 + ni * 16 + lo;
        if (row < M && col < N) {
          float v = acc[mi][ni][r];
          if (flags & GEMM_FLAG_BIAS)  v += bias[col];
          if (flags & GEMM_FLAG_RESID) v += resid[(size_t)row * ldc + col];
          C[(size_t)row * ldc + col] = v;
        }
      }
}

// ---------------------------------------------------------------------------
// Weight packing: Bt[k*ldb + n] = (n < N) ? W[n*K + k] : 0   (f32 -> bf16)
// ---------------------------------------------------------------------------
__global__ void pack_wT(const float* __restrict__ W, bf16* __restrict__ out,
                        int N, int K, int ldb) {
  int idx = blockIdx.x * blockDim.x + threadIdx.x;
  if (idx >= K * ldb) return;
  int k = idx / ldb, n = idx % ldb;
  out[idx] = (n < N) ? (bf16)W[(size_t)n * K + k] : (bf16)0.0f;
}

// patch_w (O=512, C=512, K=8) -> Bt[(tl*512 + c)][o]  (K-dim matches A rows)
__global__ void pack_patch(const float* __restrict__ W, bf16* __restrict__ out) {
  int idx = blockIdx.x * blockDim.x + threadIdx.x;
  if (idx >= 4096 * 512) return;
  int kidx = idx / 512, o = idx % 512;
  int tl = kidx / 512, c = kidx % 512;
  out[idx] = (bf16)W[((size_t)o * 512 + c) * PATCH + tl];
}

// ---------------------------------------------------------------------------
// Pre-conv (C_IN=8 -> 512, K=5, same-pad) + SiLU -> bf16  layout (b, t, 512)
// ---------------------------------------------------------------------------
__global__ void preconv_silu(const float* __restrict__ x, const float* __restrict__ w,
                             const float* __restrict__ b, bf16* __restrict__ out) {
  int idx = blockIdx.x * blockDim.x + threadIdx.x;
  if (idx >= B_SZ * T_SEQ * D_MODEL) return;
  int c = idx % D_MODEL;
  int t = (idx / D_MODEL) % T_SEQ;
  int bb = idx / (D_MODEL * T_SEQ);
  float acc = b[c];
#pragma unroll
  for (int ci = 0; ci < C_IN; ++ci)
#pragma unroll
    for (int k = 0; k < PRE_K; ++k) {
      int tt = t + k - PRE_K / 2;
      if (tt >= 0 && tt < T_SEQ)
        acc += x[((size_t)bb * T_SEQ + tt) * C_IN + ci] * w[(c * C_IN + ci) * PRE_K + k];
    }
  out[idx] = (bf16)silu_f(acc);
}

// ---------------------------------------------------------------------------
// Embed: mask-token replace + aux-MLP add; writes h_f and flipped h_b (f32)
// ---------------------------------------------------------------------------
__global__ void embed(const float* __restrict__ xtok, const int* __restrict__ mask,
                      const float* __restrict__ mask_token,
                      const float* __restrict__ y_aux, const float* __restrict__ aux_w,
                      const float* __restrict__ aux_b,
                      float* __restrict__ h_f, float* __restrict__ h_b) {
  int idx = blockIdx.x * blockDim.x + threadIdx.x;
  if (idx >= B_SZ * N_TOK * D_MODEL) return;
  int d   = idx % D_MODEL;
  int tok = (idx / D_MODEL) % N_TOK;
  int bb  = idx / (D_MODEL * N_TOK);
  float a = aux_b[d];
#pragma unroll
  for (int j = 0; j < AUX; ++j) a += y_aux[bb * AUX + j] * aux_w[d * AUX + j];
  float base = mask[bb * N_TOK + tok] ? mask_token[d]
                                      : xtok[((size_t)bb * N_TOK + tok) * D_MODEL + d];
  float v = base + silu_f(a);
  h_f[((size_t)bb * N_TOK + tok) * D_MODEL + d] = v;
  h_b[((size_t)bb * N_TOK + (N_TOK - 1 - tok)) * D_MODEL + d] = v;
}

// ---------------------------------------------------------------------------
// RMSNorm over D_MODEL=512 -> bf16; one block (256 thr) per row
// ---------------------------------------------------------------------------
__global__ __launch_bounds__(256) void rmsnorm(const float* __restrict__ h,
                                               const float* __restrict__ w,
                                               bf16* __restrict__ out) {
  __shared__ float red[256];
  int row = blockIdx.x, tid = threadIdx.x;
  float v0 = h[(size_t)row * D_MODEL + tid];
  float v1 = h[(size_t)row * D_MODEL + 256 + tid];
  red[tid] = v0 * v0 + v1 * v1;
  __syncthreads();
  for (int s = 128; s > 0; s >>= 1) {
    if (tid < s) red[tid] += red[tid + s];
    __syncthreads();
  }
  float scale = rsqrtf(red[0] / (float)D_MODEL + 1e-5f);
  out[(size_t)row * D_MODEL + tid]       = (bf16)(v0 * scale * w[tid]);
  out[(size_t)row * D_MODEL + 256 + tid] = (bf16)(v1 * scale * w[256 + tid]);
}

// ---------------------------------------------------------------------------
// Causal depthwise conv (K=4) + SiLU over xbc slice of zxbcdt; split outputs
// ---------------------------------------------------------------------------
__global__ void conv_split(const float* __restrict__ zxbcdt, const float* __restrict__ cw,
                           const float* __restrict__ cb, bf16* __restrict__ x_in,
                           bf16* __restrict__ Bm, bf16* __restrict__ Cm) {
  int idx = blockIdx.x * blockDim.x + threadIdx.x;
  if (idx >= B_SZ * N_TOK * CONV_CH) return;
  int ch = idx % CONV_CH;
  int t  = (idx / CONV_CH) % N_TOK;
  int bb = idx / (CONV_CH * N_TOK);
  float acc = cb[ch];
#pragma unroll
  for (int k = 0; k < CONV_K; ++k) {
    int tt = t - (CONV_K - 1) + k;
    if (tt >= 0)
      acc += zxbcdt[((size_t)bb * N_TOK + tt) * PROJ_PAD + D_INNER + ch] * cw[ch * CONV_K + k];
  }
  float v = silu_f(acc);
  size_t rt = (size_t)bb * N_TOK + t;
  if (ch < D_INNER)                x_in[rt * D_INNER + ch] = (bf16)v;
  else if (ch < D_INNER + D_STATE) Bm[rt * D_STATE + (ch - D_INNER)] = (bf16)v;
  else                             Cm[rt * D_STATE + (ch - D_INNER - D_STATE)] = (bf16)v;
}

// ---------------------------------------------------------------------------
// dt softplus + per-chunk cumsum of a = dt * (-exp(a_log)); 1 thr per (b,h,c)
// layouts: dtv/cum [(b*NH+h)*N_TOK + t]
// ---------------------------------------------------------------------------
__global__ void dt_cumsum(const float* __restrict__ zxbcdt, const float* __restrict__ dtb,
                          const float* __restrict__ alog, float* __restrict__ dtv,
                          float* __restrict__ cum) {
  int idx = blockIdx.x * blockDim.x + threadIdx.x;
  if (idx >= B_SZ * NH * NCHUNK) return;
  int c  = idx % NCHUNK;
  int h  = (idx / NCHUNK) % NH;
  int bb = idx / (NCHUNK * NH);
  float na  = -__expf(alog[h]);
  float run = 0.0f;
  for (int j = 0; j < CHUNK; ++j) {
    int t = c * CHUNK + j;
    float raw = zxbcdt[((size_t)bb * N_TOK + t) * PROJ_PAD + D_INNER + CONV_CH + h] + dtb[h];
    float sp  = softplus_f(raw);
    run += sp * na;
    size_t o = ((size_t)bb * NH + h) * N_TOK + t;
    dtv[o] = sp;
    cum[o] = run;
  }
}

// ---------------------------------------------------------------------------
// SSD stage A: cb[b,c][i][j] = sum_n C[i,n]*B[j,n]   (64x64x128 WMMA)
// Block per (b,c): 128 thr = 4 waves; wave w -> 16-row i-strip.
// ---------------------------------------------------------------------------
__global__ __launch_bounds__(128) void ssd_cb(const bf16* __restrict__ Cm,
                                              const bf16* __restrict__ Bm,
                                              float* __restrict__ cb) {
  __shared__ bf16 Cs[64][136];
  __shared__ bf16 BsT[128][72];
  int bc = blockIdx.x;               // b*NCHUNK + c
  int bb = bc / NCHUNK, c = bc % NCHUNK;
  int t0 = c * CHUNK;
  int tid = threadIdx.x, lane = tid & 31, wave = tid >> 5;

  {  // load C rows (row i -> 128 contiguous n)
    int i = tid >> 1, nh = tid & 1;
    const bf16* src = Cm + ((size_t)bb * N_TOK + t0 + i) * D_STATE + nh * 64;
    for (int n = 0; n < 64; ++n) Cs[i][nh * 64 + n] = src[n];
  }
  {  // load B transposed: BsT[n][j] = B[j][n]
    int j = tid >> 1, nh = tid & 1;
    const bf16* src = Bm + ((size_t)bb * N_TOK + t0 + j) * D_STATE + nh * 64;
    for (int n = 0; n < 64; ++n) BsT[nh * 64 + n][j] = src[n];
  }
  __syncthreads();

  int i0 = wave * 16;
#pragma unroll
  for (int ni = 0; ni < 4; ++ni) {
    v8f acc;
#pragma unroll
    for (int r = 0; r < 8; ++r) acc[r] = 0.0f;
#pragma unroll
    for (int ks = 0; ks < 4; ++ks) {
      v16bf a = frag_a<136>(&Cs[0][0], i0, ks * 32, lane);
      v16bf b = frag_b<72>(&BsT[0][0], ks * 32, ni * 16, lane);
      acc = wmma32(a, b, acc);
    }
    int hi = lane >> 4, lo = lane & 15;
#pragma unroll
    for (int r = 0; r < 8; ++r) {
      int i = i0 + hi * 8 + r, j = ni * 16 + lo;
      cb[(size_t)bc * 4096 + i * 64 + j] = acc[r];
    }
  }
}

// ---------------------------------------------------------------------------
// SSD stage B per (b,c,h):
//  y_intra[i][p] = sum_j (cb*seg)[i][j] * xdt[j][p]        (64x64x64 WMMA)
//  S[p][n]       = sum_j xdt[j][p]*dte[j] * B[j][n]        (64x128x64 WMMA)
// ---------------------------------------------------------------------------
__global__ __launch_bounds__(128) void ssd_intra_state(
    const bf16* __restrict__ x_in, const bf16* __restrict__ Bm,
    const float* __restrict__ cb, const float* __restrict__ dtv,
    const float* __restrict__ cum, float* __restrict__ yacc, float* __restrict__ s) {
  __shared__ bf16 Xs[64][72];    // xdt[j][p]
  __shared__ bf16 A1[64][72];    // (cb * seg)[i][j]
  __shared__ bf16 A2[64][72];    // xdt[j][p]*dte[j], transposed -> [p][j]
  __shared__ bf16 Bs2[64][136];  // B[j][n]
  __shared__ float cumS[64], dtvS[64], dteS[64];

  int bch = blockIdx.x;                      // ((b*NCHUNK+c)*NH + h)
  int h  = bch % NH;
  int bc = bch / NH;
  int c  = bc % NCHUNK, bb = bc / NCHUNK;
  int t0 = c * CHUNK;
  int tid = threadIdx.x, lane = tid & 31, wave = tid >> 5;

  if (tid < 64) {
    size_t o = ((size_t)bb * NH + h) * N_TOK + t0 + tid;
    cumS[tid] = cum[o];
    dtvS[tid] = dtv[o];
  }
  __syncthreads();

  {  // xdt, dte, B tile
    int j = tid >> 1, half = tid & 1;
    if (half == 0) dteS[j] = __expf(cumS[63] - cumS[j]);
    const bf16* xs = x_in + ((size_t)bb * N_TOK + t0 + j) * D_INNER + h * HEADDIM;
    for (int p = half * 32; p < half * 32 + 32; ++p)
      Xs[j][p] = (bf16)((float)xs[p] * dtvS[j]);
    const bf16* bs = Bm + ((size_t)bb * N_TOK + t0 + j) * D_STATE + half * 64;
    for (int n = 0; n < 64; ++n) Bs2[j][half * 64 + n] = bs[n];
  }
  {  // masked, decayed cb
    int i = tid >> 1, half = tid & 1;
    const float* cbr = cb + (size_t)bc * 4096 + i * 64;
    for (int j = half * 32; j < half * 32 + 32; ++j) {
      float v = (j <= i) ? cbr[j] * __expf(cumS[i] - cumS[j]) : 0.0f;
      A1[i][j] = (bf16)v;
    }
  }
  __syncthreads();

  {  // A2[p][j] = Xs[j][p] * dte[j]
    int p = tid >> 1, half = tid & 1;
    for (int j = half * 32; j < half * 32 + 32; ++j)
      A2[p][j] = (bf16)((float)Xs[j][p] * dteS[j]);
  }

  // y_intra (does not read A2)
  int i0 = wave * 16;
  int hi = lane >> 4, lo = lane & 15;
#pragma unroll
  for (int ni = 0; ni < 4; ++ni) {
    v8f acc;
#pragma unroll
    for (int r = 0; r < 8; ++r) acc[r] = 0.0f;
#pragma unroll
    for (int ks = 0; ks < 2; ++ks) {
      v16bf a = frag_a<72>(&A1[0][0], i0, ks * 32, lane);
      v16bf b = frag_b<72>(&Xs[0][0], ks * 32, ni * 16, lane);
      acc = wmma32(a, b, acc);
    }
#pragma unroll
    for (int r = 0; r < 8; ++r) {
      int i = i0 + hi * 8 + r, p = ni * 16 + lo;
      yacc[((size_t)bb * N_TOK + t0 + i) * D_INNER + h * HEADDIM + p] = acc[r];
    }
  }
  __syncthreads();  // A2 fully written before state WMMA

  // S[p][n]
  int p0 = wave * 16;
#pragma unroll
  for (int ni = 0; ni < 8; ++ni) {
    v8f acc;
#pragma unroll
    for (int r = 0; r < 8; ++r) acc[r] = 0.0f;
#pragma unroll
    for (int ks = 0; ks < 2; ++ks) {
      v16bf a = frag_a<72>(&A2[0][0], p0, ks * 32, lane);
      v16bf b = frag_b<136>(&Bs2[0][0], ks * 32, ni * 16, lane);
      acc = wmma32(a, b, acc);
    }
#pragma unroll
    for (int r = 0; r < 8; ++r) {
      int p = p0 + hi * 8 + r, n = ni * 16 + lo;
      s[((size_t)bch * HEADDIM + p) * D_STATE + n] = acc[r];
    }
  }
}

// ---------------------------------------------------------------------------
// Sequential inter-chunk scan per (b,h); writes hprev TRANSPOSED [n][p]
// ---------------------------------------------------------------------------
__global__ __launch_bounds__(256) void ssd_scan(const float* __restrict__ s,
                                                const float* __restrict__ cum,
                                                float* __restrict__ hprevT) {
  int bh = blockIdx.x;               // b*NH + h
  int h = bh % NH, bb = bh / NH;
  int tid = threadIdx.x;
  float hs[32];
#pragma unroll
  for (int i = 0; i < 32; ++i) hs[i] = 0.0f;
  for (int c = 0; c < NCHUNK; ++c) {
    size_t base = (((size_t)bb * NCHUNK + c) * NH + h) * (HEADDIM * D_STATE);
    float cd = __expf(cum[((size_t)bb * NH + h) * N_TOK + c * CHUNK + CHUNK - 1]);
#pragma unroll
    for (int i = 0; i < 32; ++i) {
      int e = tid + i * 256;
      int p = e >> 7, n = e & 127;
      hprevT[base + (size_t)n * HEADDIM + p] = hs[i];
      hs[i] = cd * hs[i] + s[base + (size_t)p * D_STATE + n];
    }
  }
}

// ---------------------------------------------------------------------------
// SSD stage C per (b,c,h): y_inter + combine + SiLU(z) gate -> bf16 for out_proj
//  y_inter[i][p] = exp(cum[i]) * sum_n C[i,n]*hprev[p,n]   (64x64x128 WMMA)
// ---------------------------------------------------------------------------
__global__ __launch_bounds__(128) void ssd_inter_combine(
    const bf16* __restrict__ Cm, const float* __restrict__ hprevT,
    const float* __restrict__ cum, const float* __restrict__ yacc,
    const bf16* __restrict__ x_in, const float* __restrict__ zxbcdt,
    const float* __restrict__ dskip, bf16* __restrict__ ybf) {
  __shared__ bf16 Cs[64][136];
  __shared__ bf16 Hs[128][72];   // Hs[n][p] = hprev[p][n]
  __shared__ float cumS[64];

  int bch = blockIdx.x;
  int h  = bch % NH;
  int bc = bch / NH;
  int c  = bc % NCHUNK, bb = bc / NCHUNK;
  int t0 = c * CHUNK;
  int tid = threadIdx.x, lane = tid & 31, wave = tid >> 5;

  if (tid < 64) cumS[tid] = cum[((size_t)bb * NH + h) * N_TOK + t0 + tid];
  {
    int i = tid >> 1, nh = tid & 1;
    const bf16* src = Cm + ((size_t)bb * N_TOK + t0 + i) * D_STATE + nh * 64;
    for (int n = 0; n < 64; ++n) Cs[i][nh * 64 + n] = src[n];
  }
  {
    size_t base = (size_t)bch * (HEADDIM * D_STATE);
    int n = tid;  // needs 128 threads
    const float* src = hprevT + base + (size_t)n * HEADDIM;
    for (int p = 0; p < HEADDIM; ++p) Hs[n][p] = (bf16)src[p];
  }
  __syncthreads();

  float dsk = dskip[h];
  int i0 = wave * 16;
  int hi = lane >> 4, lo = lane & 15;
#pragma unroll
  for (int ni = 0; ni < 4; ++ni) {
    v8f acc;
#pragma unroll
    for (int r = 0; r < 8; ++r) acc[r] = 0.0f;
#pragma unroll
    for (int ks = 0; ks < 4; ++ks) {
      v16bf a = frag_a<136>(&Cs[0][0], i0, ks * 32, lane);
      v16bf b = frag_b<72>(&Hs[0][0], ks * 32, ni * 16, lane);
      acc = wmma32(a, b, acc);
    }
#pragma unroll
    for (int r = 0; r < 8; ++r) {
      int i = i0 + hi * 8 + r, p = ni * 16 + lo;
      size_t rt  = (size_t)bb * N_TOK + t0 + i;
      int    col = h * HEADDIM + p;
      float y = yacc[rt * D_INNER + col] + __expf(cumS[i]) * acc[r] +
                dsk * (float)x_in[rt * D_INNER + col];
      float z = zxbcdt[rt * PROJ_PAD + col];
      ybf[rt * D_INNER + col] = (bf16)(y * silu_f(z));
    }
  }
}

// ---------------------------------------------------------------------------
// hcat: [h_f, flip(h_b)] -> bf16 (M_ROWS x 1024) for head GEMM
// ---------------------------------------------------------------------------
__global__ void hcat_pack(const float* __restrict__ h_f, const float* __restrict__ h_b,
                          bf16* __restrict__ out) {
  int idx = blockIdx.x * blockDim.x + threadIdx.x;
  if (idx >= B_SZ * N_TOK * 2 * D_MODEL) return;
  int d   = idx % (2 * D_MODEL);
  int tok = (idx / (2 * D_MODEL)) % N_TOK;
  int bb  = idx / (2 * D_MODEL * N_TOK);
  float v = (d < D_MODEL)
                ? h_f[((size_t)bb * N_TOK + tok) * D_MODEL + d]
                : h_b[((size_t)bb * N_TOK + (N_TOK - 1 - tok)) * D_MODEL + (d - D_MODEL)];
  out[idx] = (bf16)v;
}

// ---------------------------------------------------------------------------
// Un-patch + mask blend -> final output (B, T_SEQ, C_OUT) f32
// ---------------------------------------------------------------------------
__global__ void unpatch(const float* __restrict__ pp, const int* __restrict__ mask,
                        const float* __restrict__ x_seq, float* __restrict__ out) {
  int idx = blockIdx.x * blockDim.x + threadIdx.x;
  if (idx >= B_SZ * T_SEQ * C_OUT) return;
  int co = idx % C_OUT;
  int t  = (idx / C_OUT) % T_SEQ;
  int bb = idx / (C_OUT * T_SEQ);
  int tok = t / PATCH, pl = t % PATCH;
  float pv = pp[((size_t)bb * N_TOK + tok) * (C_OUT * PATCH) + co * PATCH + pl];
  out[idx] = mask[bb * N_TOK + tok] ? pv : x_seq[idx];
}

// ---------------------------------------------------------------------------
// Orchestration
// ---------------------------------------------------------------------------
static inline int cdiv(int a, int b) { return (a + b - 1) / b; }

extern "C" void kernel_launch(void* const* d_in, const int* in_sizes, int n_in,
                              void* d_out, int out_size, void* d_ws, size_t ws_size,
                              hipStream_t stream) {
  (void)in_sizes; (void)n_in; (void)out_size; (void)ws_size;
  const float* x_seq    = (const float*)d_in[0];
  const float* y_aux    = (const float*)d_in[1];
  const int*   mask     = (const int*)d_in[2];
  const float* preconvw = (const float*)d_in[3];
  const float* preconvb = (const float*)d_in[4];
  const float* patchw   = (const float*)d_in[5];
  const float* patchb   = (const float*)d_in[6];
  const float* masktok  = (const float*)d_in[7];
  const float* auxw     = (const float*)d_in[8];
  const float* auxb     = (const float*)d_in[9];
  const float* headw    = (const float*)d_in[10];
  const float* headb    = (const float*)d_in[11];
  const float* rms_a[2]   = {(const float*)d_in[12], (const float*)d_in[20]};
  const float* inw_a[2]   = {(const float*)d_in[13], (const float*)d_in[21]};
  const float* convw_a[2] = {(const float*)d_in[14], (const float*)d_in[22]};
  const float* convb_a[2] = {(const float*)d_in[15], (const float*)d_in[23]};
  const float* alog_a[2]  = {(const float*)d_in[16], (const float*)d_in[24]};
  const float* dskip_a[2] = {(const float*)d_in[17], (const float*)d_in[25]};
  const float* dtb_a[2]   = {(const float*)d_in[18], (const float*)d_in[26]};
  const float* outw_a[2]  = {(const float*)d_in[19], (const float*)d_in[27]};

  // workspace bump allocator
  char* wsb = (char*)d_ws;
  size_t off = 0;
  auto alloc = [&](size_t bytes) -> void* {
    void* p = wsb + off;
    off = (off + bytes + 255) & ~(size_t)255;
    return p;
  };
  bf16*  pre     = (bf16*)alloc((size_t)B_SZ * T_SEQ * D_MODEL * 2);
  bf16*  patchWT = (bf16*)alloc((size_t)4096 * 512 * 2);
  float* xtok    = (float*)alloc((size_t)M_ROWS * D_MODEL * 4);
  float* h_f     = (float*)alloc((size_t)M_ROWS * D_MODEL * 4);
  float* h_b     = (float*)alloc((size_t)M_ROWS * D_MODEL * 4);
  bf16*  xn      = (bf16*)alloc((size_t)M_ROWS * D_MODEL * 2);
  bf16*  inWT    = (bf16*)alloc((size_t)D_MODEL * PROJ_PAD * 2);
  float* zxbcdt  = (float*)alloc((size_t)M_ROWS * PROJ_PAD * 4);
  bf16*  x_in    = (bf16*)alloc((size_t)M_ROWS * D_INNER * 2);
  bf16*  Bm      = (bf16*)alloc((size_t)M_ROWS * D_STATE * 2);
  bf16*  Cm      = (bf16*)alloc((size_t)M_ROWS * D_STATE * 2);
  float* dtv     = (float*)alloc((size_t)B_SZ * NH * N_TOK * 4);
  float* cum     = (float*)alloc((size_t)B_SZ * NH * N_TOK * 4);
  float* cb      = (float*)alloc((size_t)B_SZ * NCHUNK * 4096 * 4);
  float* sbuf    = (float*)alloc((size_t)B_SZ * NCHUNK * NH * HEADDIM * D_STATE * 4);
  float* hprevT  = (float*)alloc((size_t)B_SZ * NCHUNK * NH * HEADDIM * D_STATE * 4);
  float* yacc    = (float*)alloc((size_t)M_ROWS * D_INNER * 4);
  bf16*  ybf     = (bf16*)alloc((size_t)M_ROWS * D_INNER * 2);
  bf16*  outWT   = (bf16*)alloc((size_t)D_INNER * D_MODEL * 2);
  bf16*  hcat    = (bf16*)alloc((size_t)M_ROWS * 2 * D_MODEL * 2);
  bf16*  headWT  = (bf16*)alloc((size_t)D_INNER * 64 * 2);
  float* pp      = (float*)alloc((size_t)M_ROWS * 64 * 4);

  // ---- stem ----
  preconv_silu<<<cdiv(B_SZ * T_SEQ * D_MODEL, 256), 256, 0, stream>>>(
      x_seq, preconvw, preconvb, pre);
  pack_patch<<<cdiv(4096 * 512, 256), 256, 0, stream>>>(patchw, patchWT);
  gemm_bf16_wmma<<<dim3(512 / 64, M_ROWS / 128), 256, 0, stream>>>(
      pre, patchWT, xtok, patchb, nullptr,
      M_ROWS, 512, 4096, 4096, 512, 512, GEMM_FLAG_BIAS);
  embed<<<cdiv(B_SZ * N_TOK * D_MODEL, 256), 256, 0, stream>>>(
      xtok, mask, masktok, y_aux, auxw, auxb, h_f, h_b);

  // ---- mamba layers, both directions ----
  for (int layer = 0; layer < N_LAYERS; ++layer) {
    for (int dir = 0; dir < 2; ++dir) {
      float* h = dir ? h_b : h_f;
      const float* rms   = rms_a[dir]   + (size_t)layer * D_MODEL;
      const float* inw   = inw_a[dir]   + (size_t)layer * PROJ * D_MODEL;
      const float* convw = convw_a[dir] + (size_t)layer * CONV_CH * CONV_K;
      const float* convb = convb_a[dir] + (size_t)layer * CONV_CH;
      const float* alog  = alog_a[dir]  + (size_t)layer * NH;
      const float* dskip = dskip_a[dir] + (size_t)layer * NH;
      const float* dtb   = dtb_a[dir]   + (size_t)layer * NH;
      const float* outw  = outw_a[dir]  + (size_t)layer * D_MODEL * D_INNER;

      rmsnorm<<<M_ROWS, 256, 0, stream>>>(h, rms, xn);
      pack_wT<<<cdiv(D_MODEL * PROJ_PAD, 256), 256, 0, stream>>>(
          inw, inWT, PROJ, D_MODEL, PROJ_PAD);
      gemm_bf16_wmma<<<dim3(cdiv(PROJ, 64), M_ROWS / 128), 256, 0, stream>>>(
          xn, inWT, zxbcdt, nullptr, nullptr,
          M_ROWS, PROJ, D_MODEL, D_MODEL, PROJ_PAD, PROJ_PAD, 0);
      conv_split<<<cdiv(B_SZ * N_TOK * CONV_CH, 256), 256, 0, stream>>>(
          zxbcdt, convw, convb, x_in, Bm, Cm);
      dt_cumsum<<<cdiv(B_SZ * NH * NCHUNK, 128), 128, 0, stream>>>(
          zxbcdt, dtb, alog, dtv, cum);
      ssd_cb<<<B_SZ * NCHUNK, 128, 0, stream>>>(Cm, Bm, cb);
      ssd_intra_state<<<B_SZ * NCHUNK * NH, 128, 0, stream>>>(
          x_in, Bm, cb, dtv, cum, yacc, sbuf);
      ssd_scan<<<B_SZ * NH, 256, 0, stream>>>(sbuf, cum, hprevT);
      ssd_inter_combine<<<B_SZ * NCHUNK * NH, 128, 0, stream>>>(
          Cm, hprevT, cum, yacc, x_in, zxbcdt, dskip, ybf);
      pack_wT<<<cdiv(D_INNER * D_MODEL, 256), 256, 0, stream>>>(
          outw, outWT, D_MODEL, D_INNER, D_MODEL);
      gemm_bf16_wmma<<<dim3(D_MODEL / 64, M_ROWS / 128), 256, 0, stream>>>(
          ybf, outWT, h, nullptr, h,
          M_ROWS, D_MODEL, D_INNER, D_INNER, D_MODEL, D_MODEL, GEMM_FLAG_RESID);
    }
  }

  // ---- head ----
  hcat_pack<<<cdiv(B_SZ * N_TOK * 2 * D_MODEL, 256), 256, 0, stream>>>(h_f, h_b, hcat);
  pack_wT<<<cdiv(D_INNER * 64, 256), 256, 0, stream>>>(headw, headWT, 64, D_INNER, 64);
  gemm_bf16_wmma<<<dim3(1, M_ROWS / 128), 256, 0, stream>>>(
      hcat, headWT, pp, headb, nullptr,
      M_ROWS, 64, D_INNER, D_INNER, 64, 64, GEMM_FLAG_BIAS);
  unpatch<<<cdiv(B_SZ * T_SEQ * C_OUT, 256), 256, 0, stream>>>(
      pp, mask, x_seq, (float*)d_out);
}
